// GraphAutoencoderNeighborhood_498216206463
// MI455X (gfx1250) — matrix operand order repaired and checked by reading, not compiled, not run on hardware
//
#include <hip/hip_runtime.h>
#include <hip/hip_bf16.h>
#include <math.h>

// ---------------------------------------------------------------------------
// GraphAutoencoderNeighborhood for MI455X (gfx1250, wave32, WMMA)
//  - dense linears: v_wmma_f32_16x16x32_f16; branch-free inner loop
//    (WT templated so every wave's column-tile group is full -> EXEC all-ones,
//     no divergent guards, accumulators stay pinned in registers)
//  - A inputs staged as f16 with K padded to /32; weights pre-packed into
//    B-fragment order so every lane loads contiguous 32B per chunk
//  - GAT softmax/aggregation via L2 atomics (arrays fit in 192MB L2)
// ---------------------------------------------------------------------------

typedef __attribute__((ext_vector_type(16))) _Float16 v16h;
typedef __attribute__((ext_vector_type(8)))  _Float16 v8h;
typedef __attribute__((ext_vector_type(8)))  float    v8f;

#define NEG_SLOPE 0.2f

static inline int cdiv(int a, int b) { return (a + b - 1) / b; }
static inline int kpad(int k) { return (k + 31) & ~31; }
static inline size_t wpack_halves(int K, int M) {
  return (size_t)cdiv(M, 16) * cdiv(K, 32) * 512;
}

// monotonic float <-> uint encoding for atomic max on floats
__device__ __forceinline__ unsigned fenc(float f) {
  unsigned u = __float_as_uint(f);
  return (u & 0x80000000u) ? ~u : (u | 0x80000000u);
}
__device__ __forceinline__ float fdec(unsigned u) {
  u = (u & 0x80000000u) ? (u & 0x7FFFFFFFu) : ~u;
  return __uint_as_float(u);
}

// ---------------------------------------------------------------------------
// Weight packer: W[K,M] f32 (row-major) -> Wp f16 in B-fragment order.
// Wp index = ((tcol*KC + kc)*32 + lane)*16 + j
//   lane = 16h + r ; element j -> k = kc*32 + 8h + j (j<8), kc*32 + 16 + 8h + (j-8) (j>=8) ; col = tcol*16 + r
// Out-of-range (k>=K || col>=M) packs as 0 so the GEMM needs no guards.
// ---------------------------------------------------------------------------
__global__ __launch_bounds__(256) void pack_w(
    const float* __restrict__ W, _Float16* __restrict__ Wp,
    int K, int M, int KC, int total)
{
  int i = blockIdx.x * blockDim.x + threadIdx.x;
  if (i >= total) return;
  int j    = i & 15;
  int lane = (i >> 4) & 31;
  int kc   = (i >> 9) % KC;
  int tcol = i / (KC << 9);
  int h = lane >> 4, r = lane & 15;
  int k = kc * 32 + 8 * h + j + ((j >> 3) << 3);
  int col = tcol * 16 + r;
  float v = (k < K && col < M) ? W[(size_t)k * M + col] : 0.0f;
  Wp[i] = (_Float16)v;
}

// ---------------------------------------------------------------------------
// WMMA GEMM: C[N,M] = act( A[N,Kp](f16, zero-padded K) @ W + bias )
// One wave handles a 16-row band x WTT column tiles.  The launcher guarantees
// tm % WTT == 0, so there are NO guards in the K loop: per chunk we issue one
// A-fragment load, WTT B-fragment loads, then WTT back-to-back WMMAs.
// Optional f32 output Cf[N,M] and/or f16 output Ch[N,Mp].
// act: 0=none, 1=relu, 2=sigmoid
// ---------------------------------------------------------------------------
template <int WTT>
__global__ __launch_bounds__(256) void wmma_gemm(
    const _Float16* __restrict__ A, const _Float16* __restrict__ Wp,
    const float* __restrict__ bias,
    float* __restrict__ Cf, _Float16* __restrict__ Ch,
    int N, int Kp, int M, int Mp, int act)
{
  int wave = (blockIdx.x * blockDim.x + threadIdx.x) >> 5;
  int lane = threadIdx.x & 31;
  int tm = (M + 15) >> 4;
  int tg = tm / WTT;                      // exact by construction
  int tn = (N + 15) >> 4;
  if (wave >= tn * tg) return;            // wave-uniform exit
  int trow  = wave / tg;
  int tcol0 = (wave % tg) * WTT;
  int h = lane >> 4, r = lane & 15;
  int KC = Kp >> 5;

  const _Float16* Ap = A + (size_t)(trow * 16 + r) * Kp + 8 * h;
  const _Float16* Bp = Wp + ((size_t)tcol0 * KC * 32 + lane) * 16;

  v8f acc[WTT];
#pragma unroll
  for (int t = 0; t < WTT; ++t) acc[t] = (v8f){};

  for (int kc = 0; kc < KC; ++kc) {
    // A fragment: two contiguous 16B runs per lane (buffer is padded)
    v8h a0 = *(const v8h*)(Ap + kc * 32);
    v8h a1 = *(const v8h*)(Ap + kc * 32 + 16);
    v16h a;
#pragma unroll
    for (int j = 0; j < 8; ++j) { a[j] = a0[j]; a[j + 8] = a1[j]; }

    // all B fragments first (one load clause), then a WMMA burst
    v16h b[WTT];
#pragma unroll
    for (int t = 0; t < WTT; ++t) {
      const _Float16* bp = Bp + ((size_t)t * KC + kc) * 512;
      v8h b0 = *(const v8h*)bp;
      v8h b1 = *(const v8h*)(bp + 8);
#pragma unroll
      for (int j = 0; j < 8; ++j) { b[t][j] = b0[j]; b[t][j + 8] = b1[j]; }
    }
#pragma unroll
    for (int t = 0; t < WTT; ++t) {
      acc[t] = __builtin_amdgcn_wmma_f32_16x16x32_f16(
          false, a, false, b[t], (short)0, acc[t], false, false);
    }
  }

#pragma unroll
  for (int t = 0; t < WTT; ++t) {
#pragma unroll
    for (int v = 0; v < 8; ++v) {
      int m = trow * 16 + v + 8 * h;
      int n = (tcol0 + t) * 16 + r;
      if (m < N && n < M) {
        float c = acc[t][v] + (bias ? bias[n] : 0.0f);
        if (act == 1)      c = fmaxf(c, 0.0f);
        else if (act == 2) c = 1.0f / (1.0f + __expf(-c));
        if (Cf) Cf[(size_t)m * M + n] = c;
        if (Ch) Ch[(size_t)m * Mp + n] = (_Float16)c;
      }
    }
  }
}

// ---------------------------------------------------------------------------
// x16[n, 0:8]=emb[ids[n]], [8:23]=x_cont[n], [23:32]=0   (f16, stride 32)
// ---------------------------------------------------------------------------
__global__ __launch_bounds__(256) void gather_x16(
    const int* __restrict__ ids, const float* __restrict__ xc,
    const float* __restrict__ emb, _Float16* __restrict__ x16, int N, int Np)
{
  int i = blockIdx.x * blockDim.x + threadIdx.x;
  if (i >= Np * 32) return;
  int n = i >> 5, j = i & 31;
  float v = 0.0f;
  if (n < N) {
    if (j < 8)       v = emb[(size_t)ids[n] * 8 + j];
    else if (j < 23) v = xc[(size_t)n * 15 + (j - 8)];
  }
  x16[i] = (_Float16)v;
}

// per-(node, head) attention scores
__global__ __launch_bounds__(256) void att_scores(
    const float* __restrict__ hbuf, const float* __restrict__ att_s,
    const float* __restrict__ att_d, float* __restrict__ asrc,
    float* __restrict__ adst, int N, int H, int C)
{
  int i = blockIdx.x * blockDim.x + threadIdx.x;
  if (i >= N * H) return;
  int n = i / H, hh = i % H;
  const float* hp = hbuf + (size_t)n * H * C + hh * C;
  const float* as = att_s + hh * C;
  const float* ad = att_d + hh * C;
  float ss = 0.f, sd = 0.f;
  for (int c = 0; c < C; ++c) { float v = hp[c]; ss += v * as[c]; sd += v * ad[c]; }
  asrc[i] = ss;
  adst[i] = sd;
}

__device__ __forceinline__ void edge_sd(const int* ei, int E, int e, int& s, int& d) {
  if (e < E) { s = ei[e]; d = ei[E + e]; }
  else       { s = d = e - E; }          // appended self loops
}

__global__ __launch_bounds__(256) void edge_max_k(
    const int* __restrict__ ei, int E, int N, int H,
    const float* __restrict__ asrc, const float* __restrict__ adst,
    unsigned* __restrict__ amax)
{
  int i = blockIdx.x * blockDim.x + threadIdx.x;
  if (i >= (E + N) * H) return;
  int e = i / H, hh = i % H;
  int s, d; edge_sd(ei, E, e, s, d);
  float al = asrc[s * H + hh] + adst[d * H + hh];
  al = (al > 0.f) ? al : NEG_SLOPE * al;
  atomicMax(&amax[d * H + hh], fenc(al));
}

__global__ __launch_bounds__(256) void edge_den_k(
    const int* __restrict__ ei, int E, int N, int H,
    const float* __restrict__ asrc, const float* __restrict__ adst,
    const unsigned* __restrict__ amax, float* __restrict__ den)
{
  int i = blockIdx.x * blockDim.x + threadIdx.x;
  if (i >= (E + N) * H) return;
  int e = i / H, hh = i % H;
  int s, d; edge_sd(ei, E, e, s, d);
  float al = asrc[s * H + hh] + adst[d * H + hh];
  al = (al > 0.f) ? al : NEG_SLOPE * al;
  atomicAdd(&den[d * H + hh], __expf(al - fdec(amax[d * H + hh])));
}

__global__ __launch_bounds__(256) void edge_agg_k(
    const int* __restrict__ ei, int E, int N, int H, int C,
    const float* __restrict__ asrc, const float* __restrict__ adst,
    const unsigned* __restrict__ amax, const float* __restrict__ den,
    const float* __restrict__ hbuf, float* __restrict__ out)
{
  int i = blockIdx.x * blockDim.x + threadIdx.x;
  if (i >= (E + N) * H) return;
  int e = i / H, hh = i % H;
  int s, d; edge_sd(ei, E, e, s, d);
  float al = asrc[s * H + hh] + adst[d * H + hh];
  al = (al > 0.f) ? al : NEG_SLOPE * al;
  float ea = __expf(al - fdec(amax[d * H + hh]));
  float a  = ea / (den[d * H + hh] + 1e-16f);
  const float* hp = hbuf + (size_t)s * H * C + hh * C;
  float*       op = out  + (size_t)d * H * C + hh * C;
  for (int c = 0; c < C; ++c) atomicAdd(&op[c], hp[c] * a);
}

// batchnorm statistics: one block per channel
__global__ __launch_bounds__(256) void bn_stats(
    const float* __restrict__ x, int N, int D,
    float* __restrict__ mean, float* __restrict__ istd)
{
  int d = blockIdx.x;
  float s = 0.f, s2 = 0.f;
  for (int n = threadIdx.x; n < N; n += blockDim.x) {
    float v = x[(size_t)n * D + d];
    s += v; s2 += v * v;
  }
  __shared__ float sh[256], sh2[256];
  sh[threadIdx.x] = s; sh2[threadIdx.x] = s2;
  __syncthreads();
  for (int st = 128; st > 0; st >>= 1) {
    if (threadIdx.x < st) {
      sh[threadIdx.x] += sh[threadIdx.x + st];
      sh2[threadIdx.x] += sh2[threadIdx.x + st];
    }
    __syncthreads();
  }
  if (threadIdx.x == 0) {
    float m = sh[0] / (float)N;
    float var = sh2[0] / (float)N - m * m;
    mean[d] = m;
    istd[d] = rsqrtf(var + 1e-5f);
  }
}

// BN + ReLU, writing the next layer's f16 A-buffer (stride Dp, zero pad)
__global__ __launch_bounds__(256) void bn_relu_f16(
    const float* __restrict__ x, int N, int Np, int D, int Dp,
    const float* __restrict__ mean, const float* __restrict__ istd,
    const float* __restrict__ g, const float* __restrict__ b,
    _Float16* __restrict__ y)
{
  int i = blockIdx.x * blockDim.x + threadIdx.x;
  if (i >= Np * Dp) return;
  int n = i / Dp, dp = i % Dp;
  float v = 0.0f;
  if (n < N && dp < D)
    v = fmaxf((x[(size_t)n * D + dp] - mean[dp]) * istd[dp] * g[dp] + b[dp], 0.0f);
  y[i] = (_Float16)v;
}

// z = mu + eps*exp(0.5*clip(lv)) -> f16 [Np,64] (pad 0); accumulate KL sum
__global__ __launch_bounds__(256) void z_kernel(
    const float* __restrict__ mu, const float* __restrict__ lv_in,
    const float* __restrict__ eps, _Float16* __restrict__ z16,
    float* __restrict__ klacc, int N, int Np)
{
  int i = blockIdx.x * blockDim.x + threadIdx.x;
  float t = 0.f;
  _Float16 zv = (_Float16)0.0f;
  bool inb = (i < Np * 64);
  int n = i >> 6, c = i & 63;
  if (inb && n < N && c < 48) {
    int idx = n * 48 + c;
    float m  = mu[idx];
    float lv = fminf(fmaxf(lv_in[idx], -20.0f), 20.0f);
    zv = (_Float16)(m + eps[idx] * __expf(0.5f * lv));
    t = 1.0f + lv - m * m - __expf(lv);
  }
  if (inb) z16[i] = zv;
  __shared__ float sh[256];
  sh[threadIdx.x] = t;
  __syncthreads();
  for (int st = 128; st > 0; st >>= 1) {
    if (threadIdx.x < st) sh[threadIdx.x] += sh[threadIdx.x + st];
    __syncthreads();
  }
  if (threadIdx.x == 0) atomicAdd(klacc, sh[0]);
}

__global__ __launch_bounds__(256) void bias_sigmoid_k(
    const float* __restrict__ x, const float* __restrict__ bias,
    float* __restrict__ out, int N, int D)
{
  int i = blockIdx.x * blockDim.x + threadIdx.x;
  if (i >= N * D) return;
  int d = i % D;
  out[i] = 1.0f / (1.0f + __expf(-(x[i] + bias[d])));
}

__global__ void kl_final_k(const float* __restrict__ klacc,
                           float* __restrict__ out, int tot)
{
  out[0] = -0.5f * klacc[0] / (float)tot;
}

// ---------------------------------------------------------------------------
// host-side helpers
// ---------------------------------------------------------------------------
static void launch_pack(const float* W, _Float16* Wp, int K, int M, hipStream_t s)
{
  int KC = kpad(K) >> 5;
  int total = (int)wpack_halves(K, M);
  pack_w<<<cdiv(total, 256), 256, 0, s>>>(W, Wp, K, M, KC, total);
}

static void launch_gemm(const _Float16* A, const _Float16* Wp, const float* bias,
                        float* Cf, _Float16* Ch,
                        int N, int Kp, int M, int Mp, int act, hipStream_t s)
{
  int tm = cdiv(M, 16);
  int wt = (tm % 4 == 0) ? 4 : (tm % 3 == 0) ? 3 : (tm % 2 == 0) ? 2 : 1;
  int waves = cdiv(N, 16) * (tm / wt);
  int blocks = cdiv(waves, 8);
  switch (wt) {
    case 4: wmma_gemm<4><<<blocks, 256, 0, s>>>(A, Wp, bias, Cf, Ch, N, Kp, M, Mp, act); break;
    case 3: wmma_gemm<3><<<blocks, 256, 0, s>>>(A, Wp, bias, Cf, Ch, N, Kp, M, Mp, act); break;
    case 2: wmma_gemm<2><<<blocks, 256, 0, s>>>(A, Wp, bias, Cf, Ch, N, Kp, M, Mp, act); break;
    default: wmma_gemm<1><<<blocks, 256, 0, s>>>(A, Wp, bias, Cf, Ch, N, Kp, M, Mp, act); break;
  }
}

static void gat_layer(const _Float16* x16, int Kp, int H, int C,
                      const _Float16* Wp, const float* att_s, const float* att_d,
                      float* hbuf, float* outbuf,
                      const int* ei, int N, int E,
                      float* asrc, float* adst, unsigned* amax, float* den,
                      hipStream_t s)
{
  int D = H * C;
  launch_gemm(x16, Wp, nullptr, hbuf, nullptr, N, Kp, D, 0, 0, s);
  att_scores<<<cdiv(N * H, 256), 256, 0, s>>>(hbuf, att_s, att_d, asrc, adst, N, H, C);
  hipMemsetAsync(amax, 0, (size_t)N * H * sizeof(unsigned), s);
  hipMemsetAsync(den, 0, (size_t)N * H * sizeof(float), s);
  hipMemsetAsync(outbuf, 0, (size_t)N * D * sizeof(float), s);
  int tot = (E + N) * H;
  edge_max_k<<<cdiv(tot, 256), 256, 0, s>>>(ei, E, N, H, asrc, adst, amax);
  edge_den_k<<<cdiv(tot, 256), 256, 0, s>>>(ei, E, N, H, asrc, adst, amax, den);
  edge_agg_k<<<cdiv(tot, 256), 256, 0, s>>>(ei, E, N, H, C, asrc, adst, amax, den, hbuf, outbuf);
}

extern "C" void kernel_launch(void* const* d_in, const int* in_sizes, int n_in,
                              void* d_out, int out_size, void* d_ws, size_t ws_size,
                              hipStream_t stream)
{
  const int N  = in_sizes[0];
  const int E  = in_sizes[2] / 2;
  const int Np = cdiv(N, 16) * 16;

  const int*   node_ids = (const int*)d_in[0];
  const float* x_cont   = (const float*)d_in[1];
  const int*   ei       = (const int*)d_in[2];
  const float* eps      = (const float*)d_in[3];

  // params in JAX tree order (dict keys sorted alphabetically)
  const float* canid_b  = (const float*)d_in[4];
  const float* canid_w  = (const float*)d_in[5];
  const float* dec0_ad  = (const float*)d_in[6];
  const float* dec0_as  = (const float*)d_in[7];
  const float* dec0_lin = (const float*)d_in[9];
  const float* dec1_ad  = (const float*)d_in[10];
  const float* dec1_as  = (const float*)d_in[11];
  const float* dec1_lin = (const float*)d_in[13];
  const float* dec2_ad  = (const float*)d_in[14];
  const float* dec2_as  = (const float*)d_in[15];
  const float* dec2_b   = (const float*)d_in[16];
  const float* dec2_lin = (const float*)d_in[17];
  const float* dbn0_b   = (const float*)d_in[18];
  const float* dbn0_g   = (const float*)d_in[19];
  const float* dbn1_b   = (const float*)d_in[20];
  const float* dbn1_g   = (const float*)d_in[21];
  const float* emb      = (const float*)d_in[22];
  const float* enc0_ad  = (const float*)d_in[23];
  const float* enc0_as  = (const float*)d_in[24];
  const float* enc0_lin = (const float*)d_in[26];
  const float* enc1_ad  = (const float*)d_in[27];
  const float* enc1_as  = (const float*)d_in[28];
  const float* enc1_lin = (const float*)d_in[30];
  const float* enc2_ad  = (const float*)d_in[31];
  const float* enc2_as  = (const float*)d_in[32];
  const float* enc2_lin = (const float*)d_in[34];
  const float* ebn0_b   = (const float*)d_in[35];
  const float* ebn0_g   = (const float*)d_in[36];
  const float* ebn1_b   = (const float*)d_in[37];
  const float* ebn1_g   = (const float*)d_in[38];
  const float* ebn2_b   = (const float*)d_in[39];
  const float* ebn2_g   = (const float*)d_in[40];
  const float* mlp0_b   = (const float*)d_in[41];
  const float* mlp0_w   = (const float*)d_in[42];
  const float* mlp1_b   = (const float*)d_in[43];
  const float* mlp1_w   = (const float*)d_in[44];
  const float* mlp2_b   = (const float*)d_in[45];
  const float* mlp2_w   = (const float*)d_in[46];
  const float* zlv_b    = (const float*)d_in[47];
  const float* zlv_w    = (const float*)d_in[48];
  const float* zm_b     = (const float*)d_in[49];
  const float* zm_w     = (const float*)d_in[50];

  // ---- workspace carve-up -------------------------------------------------
  float* fb0  = (float*)d_ws;                 // [Np,256] agg / scratch
  float* fb1  = fb0 + (size_t)Np * 256;       // [Np,256] GAT h / mu,lv
  float* asrc = fb1 + (size_t)Np * 256;       // [N,4]
  float* adst = asrc + (size_t)N * 4;
  unsigned* amax = (unsigned*)(adst + (size_t)N * 4);
  float* den  = (float*)(amax + (size_t)N * 4);
  float* bnm  = den + (size_t)N * 4;          // [256]
  float* bnv  = bnm + 256;                    // [256]
  float* klacc = bnv + 256;                   // [1] (+pad)
  _Float16* xa = (_Float16*)(klacc + 16);     // [Np,256] f16 A-buffer
  _Float16* za = xa + (size_t)Np * 256;       // [Np,64]  f16 z
  _Float16* wpA = za + (size_t)Np * 64;       // packed-weight arena
  _Float16* ha = xa;                          // MLP hidden1 (xa idle then)
  _Float16* hb = xa + (size_t)Np * 64;        // MLP hidden2

  _Float16* wp = wpA;
  _Float16 *wp_enc0 = wp;  wp += wpack_halves(23, 256);
  _Float16 *wp_enc1 = wp;  wp += wpack_halves(256, 128);
  _Float16 *wp_enc2 = wp;  wp += wpack_halves(128, 96);
  _Float16 *wp_zm   = wp;  wp += wpack_halves(96, 48);
  _Float16 *wp_zlv  = wp;  wp += wpack_halves(96, 48);
  _Float16 *wp_can  = wp;  wp += wpack_halves(48, 2048);
  _Float16 *wp_m0   = wp;  wp += wpack_halves(48, 48);
  _Float16 *wp_m1   = wp;  wp += wpack_halves(48, 48);
  _Float16 *wp_m2   = wp;  wp += wpack_halves(48, 2048);
  _Float16 *wp_d0   = wp;  wp += wpack_halves(48, 96);
  _Float16 *wp_d1   = wp;  wp += wpack_halves(96, 128);
  _Float16 *wp_d2   = wp;  wp += wpack_halves(128, 15);

  float* out      = (float*)d_out;
  float* canidOut = out + (size_t)N * 15;
  float* neighOut = canidOut + (size_t)N * 2048;
  float* klOut    = out + (out_size - 1);

  // ---- pack all weights (tiny, independent) -------------------------------
  launch_pack(enc0_lin, wp_enc0, 23, 256, stream);
  launch_pack(enc1_lin, wp_enc1, 256, 128, stream);
  launch_pack(enc2_lin, wp_enc2, 128, 96, stream);
  launch_pack(zm_w,     wp_zm,   96, 48, stream);
  launch_pack(zlv_w,    wp_zlv,  96, 48, stream);
  launch_pack(canid_w,  wp_can,  48, 2048, stream);
  launch_pack(mlp0_w,   wp_m0,   48, 48, stream);
  launch_pack(mlp1_w,   wp_m1,   48, 48, stream);
  launch_pack(mlp2_w,   wp_m2,   48, 2048, stream);
  launch_pack(dec0_lin, wp_d0,   48, 96, stream);
  launch_pack(dec1_lin, wp_d1,   96, 128, stream);
  launch_pack(dec2_lin, wp_d2,   128, 15, stream);

  // ---- input gather -> f16 [Np,32] ---------------------------------------
  gather_x16<<<cdiv(Np * 32, 256), 256, 0, stream>>>(node_ids, x_cont, emb, xa, N, Np);

  // ---- encoder (GAT bias cancels under BatchNorm; skipped) ----------------
  gat_layer(xa, 32, 4, 64, wp_enc0, enc0_as, enc0_ad, fb1, fb0,
            ei, N, E, asrc, adst, amax, den, stream);
  bn_stats<<<256, 256, 0, stream>>>(fb0, N, 256, bnm, bnv);
  bn_relu_f16<<<cdiv(Np * 256, 256), 256, 0, stream>>>(fb0, N, Np, 256, 256, bnm, bnv, ebn0_g, ebn0_b, xa);

  gat_layer(xa, 256, 1, 128, wp_enc1, enc1_as, enc1_ad, fb1, fb0,
            ei, N, E, asrc, adst, amax, den, stream);
  bn_stats<<<128, 256, 0, stream>>>(fb0, N, 128, bnm, bnv);
  bn_relu_f16<<<cdiv(Np * 128, 256), 256, 0, stream>>>(fb0, N, Np, 128, 128, bnm, bnv, ebn1_g, ebn1_b, xa);

  gat_layer(xa, 128, 1, 96, wp_enc2, enc2_as, enc2_ad, fb1, fb0,
            ei, N, E, asrc, adst, amax, den, stream);
  bn_stats<<<96, 256, 0, stream>>>(fb0, N, 96, bnm, bnv);
  bn_relu_f16<<<cdiv(Np * 96, 256), 256, 0, stream>>>(fb0, N, Np, 96, 96, bnm, bnv, ebn2_g, ebn2_b, xa);

  // ---- latent -------------------------------------------------------------
  float* mu = fb1;
  float* lv = fb1 + (size_t)N * 48;
  launch_gemm(xa, wp_zm,  zm_b,  mu, nullptr, N, 96, 48, 0, 0, stream);
  launch_gemm(xa, wp_zlv, zlv_b, lv, nullptr, N, 96, 48, 0, 0, stream);
  hipMemsetAsync(klacc, 0, sizeof(float), stream);
  z_kernel<<<cdiv(Np * 64, 256), 256, 0, stream>>>(mu, lv, eps, za, klacc, N, Np);

  // ---- canid logits: [N,48]x[48,2048] (dominant GEMM) ---------------------
  launch_gemm(za, wp_can, canid_b, canidOut, nullptr, N, 64, 2048, 0, 0, stream);

  // ---- neighborhood MLP (f16 chained hidden layers) -----------------------
  hipMemsetAsync(ha, 0, (size_t)Np * 128 * sizeof(_Float16), stream); // ha+hb
  launch_gemm(za, wp_m0, mlp0_b, nullptr, ha, N, 64, 48, 64, 1, stream);
  launch_gemm(ha, wp_m1, mlp1_b, nullptr, hb, N, 64, 48, 64, 1, stream);
  launch_gemm(hb, wp_m2, mlp2_b, neighOut, nullptr, N, 64, 2048, 0, 0, stream);

  // ---- decoder ------------------------------------------------------------
  gat_layer(za, 64, 4, 24, wp_d0, dec0_as, dec0_ad, fb1, fb0,
            ei, N, E, asrc, adst, amax, den, stream);
  bn_stats<<<96, 256, 0, stream>>>(fb0, N, 96, bnm, bnv);
  bn_relu_f16<<<cdiv(Np * 96, 256), 256, 0, stream>>>(fb0, N, Np, 96, 96, bnm, bnv, dbn0_g, dbn0_b, xa);

  gat_layer(xa, 96, 4, 32, wp_d1, dec1_as, dec1_ad, fb1, fb0,
            ei, N, E, asrc, adst, amax, den, stream);
  bn_stats<<<128, 256, 0, stream>>>(fb0, N, 128, bnm, bnv);
  bn_relu_f16<<<cdiv(Np * 128, 256), 256, 0, stream>>>(fb0, N, Np, 128, 128, bnm, bnv, dbn1_g, dbn1_b, xa);

  gat_layer(xa, 128, 1, 15, wp_d2, dec2_as, dec2_ad, fb1, fb0,
            ei, N, E, asrc, adst, amax, den, stream);
  bias_sigmoid_k<<<cdiv(N * 15, 256), 256, 0, stream>>>(fb0, dec2_b, out, N, 15);

  // ---- KL scalar ----------------------------------------------------------
  kl_final_k<<<1, 1, 0, stream>>>(klacc, klOut, N * 48);
}
